// MultiheadAttention_49340584296991
// MI455X (gfx1250) — matrix operand order repaired and checked by reading, not compiled, not run on hardware
//
#include <hip/hip_runtime.h>
#include <hip/hip_bf16.h>
#include <stdint.h>

// MHA forward for B=2,S=2048,H=16,DH=64,D=1024 on gfx1250.
// All GEMMs via V_WMMA_F32_16X16X32_BF16 (bf16 in, f32 accum).
// Workspace usage: ~56MB of bf16 staging buffers.

#define S_LEN   2048
#define BATCH   2
#define HEADS   16
#define DHEAD   64
#define DMODEL  1024
#define MROWS   (BATCH * S_LEN)   // 4096

typedef __attribute__((ext_vector_type(16))) __bf16 v16bf;
typedef __attribute__((ext_vector_type(8)))  __bf16 v8bf;
typedef __attribute__((ext_vector_type(8)))  float  v8f;

// ---------------------------------------------------------------------------
// Fragment loaders (wave32 WMMA layouts, CDNA5 ISA §7.12.2)
//
// A (16x32 bf16, row-major source, lane holds row M=lane&15):
//   half = lane>>4 ; VGPR r (r<4) -> K = half*8 + 2r,2r+1
//                    VGPR r (r>=4)-> K = 16 + half*8 + 2(r-4),...
//   => two contiguous 8-element (16B) loads at kk+half*8 and kk+16+half*8.
__device__ __forceinline__ v16bf frag_a_bf16(const __bf16* __restrict__ base,
                                             int row, int ld, int kk, int lane) {
  const int half = lane >> 4;
  const __bf16* p = base + (size_t)row * ld + kk + half * 8;
  v8bf lo = *(const v8bf*)(p);
  v8bf hi = *(const v8bf*)(p + 16);
  v16bf f;
#pragma unroll
  for (int i = 0; i < 8; ++i) { f[i] = lo[i]; f[i + 8] = hi[i]; }
  return f;
}

// Same A layout but converting from f32 source in-register.
__device__ __forceinline__ v16bf frag_a_f32(const float* __restrict__ base,
                                            int row, int ld, int kk, int lane) {
  const int half = lane >> 4;
  const float* p = base + (size_t)row * ld + kk + half * 8;
  v16bf f;
#pragma unroll
  for (int i = 0; i < 8; ++i) { f[i] = (__bf16)p[i]; f[i + 8] = (__bf16)p[i + 16]; }
  return f;
}

// B (32x16 bf16). Source supplied transposed (N x K row-major): lane holds
// column N=lane&15 == a row of the transposed source. Same K order as A:
// elements 0-7 hold K = half*8 + 0..7, elements 8-15 hold K = 16 + half*8 + 0..7,
// i.e. two contiguous 8-element (16B) loads at kk+half*8 and kk+16+half*8.
__device__ __forceinline__ v16bf frag_b_bf16(const __bf16* __restrict__ base,
                                             int row, int ld, int kk, int lane) {
  const int half = lane >> 4;
  const __bf16* p = base + (size_t)row * ld + kk + half * 8;
  v8bf lo = *(const v8bf*)(p);
  v8bf hi = *(const v8bf*)(p + 16);
  v16bf f;
#pragma unroll
  for (int i = 0; i < 8; ++i) { f[i] = lo[i]; f[i + 8] = hi[i]; }
  return f;
}

#define WMMA_BF16(a, b, c) \
  __builtin_amdgcn_wmma_f32_16x16x32_bf16(false, (a), false, (b), (short)0, (c), false, false)

// ---------------------------------------------------------------------------
// Prep kernels
__global__ void k_transpose_w(const float* __restrict__ src,  // (K x N) row-major
                              __bf16* __restrict__ dst,       // (N x K) row-major
                              int K, int N) {
  int idx = blockIdx.x * blockDim.x + threadIdx.x;
  if (idx >= K * N) return;
  int n = idx / K, k = idx % K;
  dst[idx] = (__bf16)src[(size_t)k * N + n];
}

__global__ void k_cvt_bf16(const float* __restrict__ src, __bf16* __restrict__ dst, int n) {
  int idx = blockIdx.x * blockDim.x + threadIdx.x;
  if (idx < n) dst[idx] = (__bf16)src[idx];
}

// ---------------------------------------------------------------------------
// QKV projection: P = Xbf(M=4096 x K=1024) @ Wt^T(N=1024 x K=1024) + bias.
// One wave computes one 16x16 tile. mode 0: write (B,H,S,DH); mode 1: write
// V transposed (B,H,DH,S) so the AV GEMM's B fragments are contiguous.
__global__ void k_proj_gemm(const __bf16* __restrict__ X,
                            const __bf16* __restrict__ Wt,
                            const float*  __restrict__ bias,
                            __bf16* __restrict__ dstQK,
                            __bf16* __restrict__ dstVT,
                            int mode) {
  const int lane = threadIdx.x & 31;
  const int w    = blockIdx.x * 4 + (threadIdx.x >> 5);
  const int tm   = w >> 6;    // 256 M-tiles
  const int tn   = w & 63;    // 64 N-tiles
  const int rowA = tm * 16 + (lane & 15);
  const int rowB = tn * 16 + (lane & 15);
  v8f c = {};
  for (int kk = 0; kk < DMODEL; kk += 32) {
    v16bf a = frag_a_bf16(X,  rowA, DMODEL, kk, lane);
    v16bf b = frag_b_bf16(Wt, rowB, DMODEL, kk, lane);
    c = WMMA_BF16(a, b, c);
  }
  const int half = lane >> 4;
  const int n    = tn * 16 + (lane & 15);
  const int h    = n >> 6;
  const int dk   = n & 63;
  const float bs = bias[n];
#pragma unroll
  for (int r = 0; r < 8; ++r) {
    int m  = tm * 16 + r + 8 * half;
    int bb = m >> 11;              // /2048
    int q  = m & (S_LEN - 1);
    float val = c[r] + bs;
    if (mode == 0)
      dstQK[((size_t)(bb * HEADS + h) * S_LEN + q) * DHEAD + dk] = (__bf16)val;
    else
      dstVT[((size_t)(bb * HEADS + h) * DHEAD + dk) * S_LEN + q] = (__bf16)val;
  }
}

// ---------------------------------------------------------------------------
// logits[b,h,q,j] = dm * (Q.K^T) * scale + (mask ? 0 : -1e9), written f32.
__global__ void k_attn_logits(const __bf16*  __restrict__ Qb,   // (B,H,S,DH)
                              const __bf16*  __restrict__ Kb,   // (B,H,S,DH)
                              const uint8_t* __restrict__ dmask,// (B,H,S,S) bool
                              const uint8_t* __restrict__ amask,// (B,1,S,S) bool
                              float* __restrict__ attn) {       // (B,H,S,S)
  const int lane = threadIdx.x & 31;
  const int bh   = blockIdx.y;
  const int w    = blockIdx.x * 4 + (threadIdx.x >> 5);  // 16384 tiles/bh
  const int tq   = w >> 7;
  const int tj   = w & 127;
  const __bf16* Q = Qb + (size_t)bh * S_LEN * DHEAD;
  const __bf16* K = Kb + (size_t)bh * S_LEN * DHEAD;
  const int rowA = tq * 16 + (lane & 15);
  const int rowB = tj * 16 + (lane & 15);
  v8f c = {};
#pragma unroll
  for (int kk = 0; kk < DHEAD; kk += 32) {
    v16bf a = frag_a_bf16(Q, rowA, DHEAD, kk, lane);
    v16bf b = frag_b_bf16(K, rowB, DHEAD, kk, lane);
    c = WMMA_BF16(a, b, c);
  }
  const int   half   = lane >> 4;
  const int   j      = tj * 16 + (lane & 15);
  const int   bb     = bh >> 4;
  const size_t dmB   = (size_t)bh * S_LEN * S_LEN;
  const size_t mskB  = (size_t)bb * S_LEN * S_LEN;
  const float scale  = 0.125f;   // DH^-0.5
#pragma unroll
  for (int r = 0; r < 8; ++r) {
    int q = tq * 16 + r + 8 * half;
    size_t idx = dmB + (size_t)q * S_LEN + j;
    float dm = dmask[idx] ? 1.0f : 0.0f;
    float mv = amask[mskB + (size_t)q * S_LEN + j] ? 0.0f : -1e9f;
    attn[idx] = dm * c[r] * scale + mv;
  }
}

// ---------------------------------------------------------------------------
// Row softmax over the last axis (2048), in place. One 256-thread block/row.
__global__ void k_softmax(float* __restrict__ attn) {
  const size_t row = blockIdx.x;            // B*H*S = 65536 rows
  float* p = attn + row * S_LEN;
  const int t = threadIdx.x;
  __shared__ float red[256];
  float v[8];
  float m = -3.0e38f;
#pragma unroll
  for (int i = 0; i < 8; ++i) { v[i] = p[t + 256 * i]; m = fmaxf(m, v[i]); }
  red[t] = m; __syncthreads();
  for (int s = 128; s > 0; s >>= 1) { if (t < s) red[t] = fmaxf(red[t], red[t + s]); __syncthreads(); }
  m = red[0]; __syncthreads();
  float sum = 0.0f;
#pragma unroll
  for (int i = 0; i < 8; ++i) { v[i] = __expf(v[i] - m); sum += v[i]; }
  red[t] = sum; __syncthreads();
  for (int s = 128; s > 0; s >>= 1) { if (t < s) red[t] += red[t + s]; __syncthreads(); }
  float inv = 1.0f / red[0];
#pragma unroll
  for (int i = 0; i < 8; ++i) p[t + 256 * i] = v[i] * inv;
}

// ---------------------------------------------------------------------------
// preproj = attn(S x S, f32) @ V ; A converted to bf16 in-register,
// B = Vt (B,H,DH,S) bf16 rows are contiguous K.
__global__ void k_attn_v(const float*  __restrict__ attn, // (B,H,S,S)
                         const __bf16* __restrict__ Vt,   // (B,H,DH,S)
                         __bf16* __restrict__ pp) {       // (B,S,H*DH)
  const int lane = threadIdx.x & 31;
  const int bh   = blockIdx.y;
  const int w    = blockIdx.x * 4 + (threadIdx.x >> 5);  // 512 tiles/bh
  const int tq   = w >> 2;
  const int td   = w & 3;
  const float*  A  = attn + (size_t)bh * S_LEN * S_LEN;
  const __bf16* Bv = Vt   + (size_t)bh * DHEAD * S_LEN;
  const int rowA = tq * 16 + (lane & 15);
  const int rowB = td * 16 + (lane & 15);
  v8f c = {};
  for (int kk = 0; kk < S_LEN; kk += 32) {
    v16bf a = frag_a_f32(A, rowA, S_LEN, kk, lane);
    v16bf b = frag_b_bf16(Bv, rowB, S_LEN, kk, lane);
    c = WMMA_BF16(a, b, c);
  }
  const int half = lane >> 4;
  const int d    = td * 16 + (lane & 15);
  const int bb   = bh >> 4, h = bh & 15;
#pragma unroll
  for (int r = 0; r < 8; ++r) {
    int q = tq * 16 + r + 8 * half;
    pp[(size_t)(bb * S_LEN + q) * DMODEL + h * DHEAD + d] = (__bf16)c[r];
  }
}

// ---------------------------------------------------------------------------
// output = preproj(4096 x 1024) @ wo + bo, f32 result.
__global__ void k_out_proj(const __bf16* __restrict__ pp,
                           const __bf16* __restrict__ Wot, // (D x H*DH) N x K
                           const float*  __restrict__ bo,
                           float* __restrict__ out) {
  const int lane = threadIdx.x & 31;
  const int w    = blockIdx.x * 4 + (threadIdx.x >> 5);
  const int tm   = w >> 6;
  const int tn   = w & 63;
  const int rowA = tm * 16 + (lane & 15);
  const int rowB = tn * 16 + (lane & 15);
  v8f c = {};
  for (int kk = 0; kk < DMODEL; kk += 32) {
    v16bf a = frag_a_bf16(pp,  rowA, DMODEL, kk, lane);
    v16bf b = frag_b_bf16(Wot, rowB, DMODEL, kk, lane);
    c = WMMA_BF16(a, b, c);
  }
  const int half = lane >> 4;
  const int n    = tn * 16 + (lane & 15);
  const float bs = bo[n];
#pragma unroll
  for (int r = 0; r < 8; ++r) {
    int m = tm * 16 + r + 8 * half;
    out[(size_t)m * DMODEL + n] = c[r] + bs;
  }
}

// ---------------------------------------------------------------------------
extern "C" void kernel_launch(void* const* d_in, const int* in_sizes, int n_in,
                              void* d_out, int out_size, void* d_ws, size_t ws_size,
                              hipStream_t stream) {
  (void)in_sizes; (void)n_in; (void)out_size; (void)ws_size;
  const float*   q_in  = (const float*)d_in[0];   // (B,S,D)
  const float*   kv_in = (const float*)d_in[1];   // (B,S,D)
  const uint8_t* amask = (const uint8_t*)d_in[2]; // (B,1,S,S) bool
  const uint8_t* dmask = (const uint8_t*)d_in[3]; // (B,H,S,S) bool
  const float*   wq = (const float*)d_in[4];
  const float*   bq = (const float*)d_in[5];
  const float*   wk = (const float*)d_in[6];
  const float*   bk = (const float*)d_in[7];
  const float*   wv = (const float*)d_in[8];
  const float*   bv = (const float*)d_in[9];
  const float*   wo = (const float*)d_in[10];
  const float*   bo = (const float*)d_in[11];

  float* out_f  = (float*)d_out;                                   // (B,S,D)
  float* attn_f = (float*)d_out + (size_t)BATCH * S_LEN * DMODEL;  // (B,H,S,S)

  // Workspace carve-up (bf16 elements), ~56MB total.
  __bf16* Wqt = (__bf16*)d_ws;
  __bf16* Wkt = Wqt + (size_t)DMODEL * DMODEL;
  __bf16* Wvt = Wkt + (size_t)DMODEL * DMODEL;
  __bf16* Wot = Wvt + (size_t)DMODEL * DMODEL;
  __bf16* Xq  = Wot + (size_t)DMODEL * DMODEL;            // 4096*1024
  __bf16* Xkv = Xq  + (size_t)MROWS * DMODEL;
  __bf16* Qb  = Xkv + (size_t)MROWS * DMODEL;             // (B,H,S,DH)
  __bf16* Kb  = Qb  + (size_t)BATCH * HEADS * S_LEN * DHEAD;
  __bf16* Vt  = Kb  + (size_t)BATCH * HEADS * S_LEN * DHEAD;
  __bf16* PP  = Vt  + (size_t)BATCH * HEADS * S_LEN * DHEAD;

  // 1) weight transpose+convert, activation convert
  {
    const int n = DMODEL * DMODEL;
    dim3 g((n + 255) / 256), b(256);
    k_transpose_w<<<g, b, 0, stream>>>(wq, Wqt, DMODEL, DMODEL);
    k_transpose_w<<<g, b, 0, stream>>>(wk, Wkt, DMODEL, DMODEL);
    k_transpose_w<<<g, b, 0, stream>>>(wv, Wvt, DMODEL, DMODEL);
    k_transpose_w<<<g, b, 0, stream>>>(wo, Wot, DMODEL, DMODEL);
  }
  {
    const int n = MROWS * DMODEL;
    dim3 g((n + 255) / 256), b(256);
    k_cvt_bf16<<<g, b, 0, stream>>>(q_in,  Xq,  n);
    k_cvt_bf16<<<g, b, 0, stream>>>(kv_in, Xkv, n);
  }

  // 2) Q/K/V projections: 256x64 tiles, 4 waves per block -> 4096 blocks
  {
    dim3 g(4096), b(128);
    k_proj_gemm<<<g, b, 0, stream>>>(Xq,  Wqt, bq, Qb, nullptr, 0);
    k_proj_gemm<<<g, b, 0, stream>>>(Xkv, Wkt, bk, Kb, nullptr, 0);
    k_proj_gemm<<<g, b, 0, stream>>>(Xkv, Wvt, bv, nullptr, Vt, 1);
  }

  // 3) masked/dropout logits: 128x128 tiles per (b,h)
  k_attn_logits<<<dim3(4096, BATCH * HEADS), dim3(128), 0, stream>>>(
      Qb, Kb, dmask, amask, attn_f);

  // 4) row softmax, in place (also the attn_weights output)
  k_softmax<<<dim3(BATCH * HEADS * S_LEN), dim3(256), 0, stream>>>(attn_f);

  // 5) attn @ V : 128x4 tiles per (b,h)
  k_attn_v<<<dim3(128, BATCH * HEADS), dim3(128), 0, stream>>>(attn_f, Vt, PP);

  // 6) output projection
  k_out_proj<<<dim3(4096), dim3(128), 0, stream>>>(PP, Wot, bo, out_f);
}
